// QLoRALinear_50130858279248
// MI455X (gfx1250) — matrix operand order repaired and checked by reading, not compiled, not run on hardware
//
#include <hip/hip_runtime.h>
#include <hip/hip_bf16.h>

typedef __attribute__((ext_vector_type(16))) _Float16 v16h;
typedef __attribute__((ext_vector_type(8)))  _Float16 v8h;
typedef __attribute__((ext_vector_type(8)))  float    v8f;
typedef __attribute__((ext_vector_type(4)))  int      v4i;

#define D_IN   4096
#define D_OUT  4096
#define M_TOT  8192   // B*S = 4*2048
#define RANK   16
#define BK     32
#define LDA    40     // padded LDS row stride in halves (80B, spreads banks)

// ---------------------------------------------------------------------------
// CDNA5 async global->LDS copy (ASYNCcnt-tracked, no VGPR staging)
// builtin signature: (int4 AS1* src, int4 AS3* dst, imm offset, imm cpol)
// ---------------------------------------------------------------------------
__device__ __forceinline__ void async_cp16(const _Float16* g, _Float16* l) {
#if __has_builtin(__builtin_amdgcn_global_load_async_to_lds_b128)
    v4i* gg = (v4i*)const_cast<_Float16*>(g);
    v4i* ll = (v4i*)l;
    __builtin_amdgcn_global_load_async_to_lds_b128(
        (__attribute__((address_space(1))) v4i*)gg,
        (__attribute__((address_space(3))) v4i*)ll, 0, 0);
#else
    unsigned loff = (unsigned)(unsigned long long)
        (__attribute__((address_space(3))) _Float16*)l;
    asm volatile("global_load_async_to_lds_b128 %0, %1, off"
                 :: "v"(loff), "v"(g) : "memory");
#endif
}

__device__ __forceinline__ void wait_async0() {
#if __has_builtin(__builtin_amdgcn_s_wait_asynccnt)
    __builtin_amdgcn_s_wait_asynccnt(0);
#else
    asm volatile("s_wait_asynccnt 0x0" ::: "memory");
#endif
}

// ---------------------------------------------------------------------------
// Kernel 1: W_eff[n][k] = (float)qweight[n][k]*scale + SCALING*sum_r A[k][r]*B[r][n]
// int4 values are exact in f16; LoRA folded into the weight. One block per n.
// ---------------------------------------------------------------------------
__global__ __launch_bounds__(256) void qlora_build_weff(
    const int* __restrict__ qw, const float* __restrict__ scales,
    const float* __restrict__ lora_A, const float* __restrict__ lora_B,
    _Float16* __restrict__ weff)
{
    __shared__ float sBl[RANK];
    const int n = blockIdx.x;
    const int t = threadIdx.x;
    if (t < RANK) sBl[t] = lora_B[t * D_OUT + n] * 1.0f;  // SCALING = alpha/rank = 1.0
    __syncthreads();
    const float scale = scales[0];
    #pragma unroll
    for (int i = 0; i < D_IN / 256; ++i) {
        const int k = t + i * 256;
        float acc = (float)qw[(size_t)n * D_IN + k] * scale;
        #pragma unroll
        for (int r = 0; r < RANK; ++r) acc += lora_A[k * RANK + r] * sBl[r];
        weff[(size_t)n * D_IN + k] = (_Float16)acc;
    }
}

__device__ __forceinline__ v8h cvt8(float4 a, float4 b) {
    v8h h;
    h[0] = (_Float16)a.x; h[1] = (_Float16)a.y; h[2] = (_Float16)a.z; h[3] = (_Float16)a.w;
    h[4] = (_Float16)b.x; h[5] = (_Float16)b.y; h[6] = (_Float16)b.z; h[7] = (_Float16)b.w;
    return h;
}

// ---------------------------------------------------------------------------
// Kernel 1b: one-pass x fp32 -> f16 (enables pure byte-move async DMA in GEMM)
// ---------------------------------------------------------------------------
__global__ __launch_bounds__(256) void qlora_cvt_x(
    const float* __restrict__ x, _Float16* __restrict__ xh)
{
    const size_t i = ((size_t)blockIdx.x * 256 + threadIdx.x) * 8;
    float4 a = *(const float4*)(x + i);
    float4 b = *(const float4*)(x + i + 4);
    *(v8h*)(xh + i) = cvt8(a, b);
}

// ---------------------------------------------------------------------------
// Kernel 2: out[m][n] = sum_k xh[m][k] * W_eff[n][k] + bias[n]
// 128x128 block tile, BK=32, 8 waves (4 M x 2 N), wave tile 32x64.
// Async global->LDS double buffering; 8x v_wmma_f32_16x16x32_f16 per chunk.
// ---------------------------------------------------------------------------
__global__ __launch_bounds__(256) void qlora_gemm_wmma(
    const _Float16* __restrict__ xh, const _Float16* __restrict__ weff,
    const float* __restrict__ bias, float* __restrict__ out)
{
    __shared__ _Float16 sA[2][128 * LDA];
    __shared__ _Float16 sB[2][128 * LDA];

    const int t     = threadIdx.x;
    const int lane  = t & 31;
    const int waveM = (t >> 5) & 3;   // 4 waves along M
    const int waveN = (t >> 5) >> 2;  // 2 waves along N

    const int bm = blockIdx.y * 128;
    const int bn = blockIdx.x * 128;

    // copy assignment: thread owns 16 halves (two b128) of one tile row
    const int ldRow = t >> 1;           // 0..127
    const int ldK   = (t & 1) * 16;     // 0 or 16

    const _Float16* gA = xh   + (size_t)(bm + ldRow) * D_IN + ldK;
    const _Float16* gB = weff + (size_t)(bn + ldRow) * D_IN + ldK;
    const int ldsOff = ldRow * LDA + ldK;

    v8f acc[2][4] = {};

    // ---- prologue: async DMA chunk 0 into buffer 0
    async_cp16(gA + 0, &sA[0][ldsOff + 0]);
    async_cp16(gA + 8, &sA[0][ldsOff + 8]);
    async_cp16(gB + 0, &sB[0][ldsOff + 0]);
    async_cp16(gB + 8, &sB[0][ldsOff + 8]);
    wait_async0();
    __syncthreads();

    // fragment addressing (ISA 7.12.2, wave32):
    // A 16x32 f16: lane row = lane%16; halves[0..7] = K kb8..kb8+7,
    //              halves[8..15] = K kb8+16..kb8+23, kb8 = (lane/16)*8
    // B 32x16 f16: lane col = lane%16; halves[0..7] = K kb8..kb8+7,
    //              halves[8..15] = K kb8+16..kb8+23  (W_eff stored [N][K])
    const int r16  = lane & 15;
    const int kb8  = (lane >> 4) * 8;
    const int kb16 = (lane >> 4) * 16;
    const int aRow = waveM * 32;
    const int bCol = waveN * 64;

    const int kChunks = D_IN / BK;  // 128
    for (int kc = 0; kc < kChunks; ++kc) {
        const int  buf     = kc & 1;
        const bool hasNext = (kc + 1 < kChunks);

        if (hasNext) {  // DMA next chunk into the other buffer (overlaps WMMAs)
            const _Float16* gA2 = gA + (kc + 1) * BK;
            const _Float16* gB2 = gB + (kc + 1) * BK;
            async_cp16(gA2 + 0, &sA[buf ^ 1][ldsOff + 0]);
            async_cp16(gA2 + 8, &sA[buf ^ 1][ldsOff + 8]);
            async_cp16(gB2 + 0, &sB[buf ^ 1][ldsOff + 0]);
            async_cp16(gB2 + 8, &sB[buf ^ 1][ldsOff + 8]);
        }

        v16h af[2], bf[4];
        #pragma unroll
        for (int mt = 0; mt < 2; ++mt) {
            const _Float16* p = &sA[buf][(aRow + mt * 16 + r16) * LDA + kb8];
            v8h lo = *(const v8h*)p;
            v8h hi = *(const v8h*)(p + 16);
            af[mt] = __builtin_shufflevector(lo, hi, 0,1,2,3,4,5,6,7,8,9,10,11,12,13,14,15);
        }
        #pragma unroll
        for (int nt = 0; nt < 4; ++nt) {
            const _Float16* p = &sB[buf][(bCol + nt * 16 + r16) * LDA + kb8];
            v8h lo = *(const v8h*)p;
            v8h hi = *(const v8h*)(p + 16);
            bf[nt] = __builtin_shufflevector(lo, hi, 0,1,2,3,4,5,6,7,8,9,10,11,12,13,14,15);
        }

        #pragma unroll
        for (int mt = 0; mt < 2; ++mt)
            #pragma unroll
            for (int nt = 0; nt < 4; ++nt)
                acc[mt][nt] = __builtin_amdgcn_wmma_f32_16x16x32_f16(
                    false, af[mt], false, bf[nt], (short)0, acc[mt][nt], false, false);

        if (hasNext) wait_async0();   // DMA has overlapped the 8 WMMAs above
        __syncthreads();
    }

    // ---- epilogue: C/D layout: element j -> M = j + 8*(lane/16), N = lane%16
    const int col     = lane & 15;
    const int rowHalf = (lane >> 4) * 8;
    #pragma unroll
    for (int nt = 0; nt < 4; ++nt) {
        const int gn = bn + bCol + nt * 16 + col;
        const float bv = bias[gn];
        #pragma unroll
        for (int mt = 0; mt < 2; ++mt) {
            const int gm0 = bm + aRow + mt * 16 + rowHalf;
            #pragma unroll
            for (int j = 0; j < 8; ++j)
                out[(size_t)(gm0 + j) * D_OUT + gn] = acc[mt][nt][j] + bv;
        }
    }
}

extern "C" void kernel_launch(void* const* d_in, const int* in_sizes, int n_in,
                              void* d_out, int out_size, void* d_ws, size_t ws_size,
                              hipStream_t stream) {
    const float* x      = (const float*)d_in[0];
    const int*   qw     = (const int*)  d_in[1];
    const float* scales = (const float*)d_in[2];
    const float* bias   = (const float*)d_in[3];
    const float* lora_A = (const float*)d_in[4];
    const float* lora_B = (const float*)d_in[5];
    float*       out    = (float*)d_out;

    // workspace: [0,32MB) W_eff f16 [D_OUT][D_IN]; [32MB,96MB) x f16 [M][D_IN]
    _Float16* weff = (_Float16*)d_ws;
    _Float16* xh   = (_Float16*)((char*)d_ws + (size_t)D_OUT * D_IN * sizeof(_Float16));

    qlora_build_weff<<<D_OUT, 256, 0, stream>>>(qw, scales, lora_A, lora_B, weff);
    qlora_cvt_x<<<(M_TOT * (size_t)D_IN / 8) / 256, 256, 0, stream>>>(x, xh);
    qlora_gemm_wmma<<<dim3(D_OUT / 128, M_TOT / 128), 256, 0, stream>>>(xh, weff, bias, out);
}
